// ResNetBKLayer_70514773066258
// MI455X (gfx1250) — matrix operand order repaired and checked by reading, not compiled, not run on hardware
//
#include <hip/hip_runtime.h>
#include <hip/hip_bf16.h>

// ---------------------------------------------------------------------------
// Types
// ---------------------------------------------------------------------------
typedef __attribute__((ext_vector_type(16))) __bf16 v16bf;
typedef __attribute__((ext_vector_type(8)))  float  v8f;
typedef unsigned int u32;
typedef __attribute__((ext_vector_type(4)))  u32    u32x4;
typedef __attribute__((ext_vector_type(4)))  float  f32x4;

union V16BF { u32x4 q[2]; v16bf v; };

#define D_MODEL   2048
#define MT        64            // rows per block (4 waves x 16)
#define NB        4             // N tiles processed concurrently per wave
#define PITCH_X   (D_MODEL + 8) // halves; +8 pad rotates LDS banks by 4/row
#define KC        256           // w1 K-chunk in halves
#define PITCH_W   (KC + 8)
#define LDS_X_HALVES (MT * PITCH_X)
#define LDS_W_HALVES (16 * NB * PITCH_W)
#define GEMM_LDS_BYTES ((LDS_X_HALVES + LDS_W_HALVES) * 2)

__device__ __forceinline__ unsigned short f2bf(float f) {
  // round-to-nearest-even fp32 -> bf16
  u32 u = __float_as_uint(f);
  u32 r = (u + 0x7fffu + ((u >> 16) & 1u)) >> 16;
  return (unsigned short)r;
}

// ---------------------------------------------------------------------------
// Kernel 1: fused  v[i] = b2 + sum_j relu(sum_k x[i,k]*w1[j,k] + b1[j]) * w2[j]
// bf16 WMMA (16x16x32), fp32 accumulate. 4 concurrent N-tiles per wave; all
// LDS fragment loads hoisted ahead of the 4 WMMAs so the scheduler can use
// partial s_wait_dscnt and overlap LDS latency with matrix issue.
// ---------------------------------------------------------------------------
__global__ void __launch_bounds__(128)
gemm_v_kernel(const float* __restrict__ x, const float* __restrict__ w1,
              const float* __restrict__ b1, const float* __restrict__ w2,
              const float* __restrict__ b2, float* __restrict__ v_out) {
  extern __shared__ unsigned short lds[];
  unsigned short* ldsX = lds;                  // [MT][PITCH_X]      bf16 as u16
  unsigned short* ldsW = lds + LDS_X_HALVES;   // [16*NB][PITCH_W]   bf16 as u16

  const int tid  = threadIdx.x;
  const int lane = tid & 31;
  const int wave = tid >> 5;        // 0..3 -> 16-row M sub-tile
  const int hi   = (lane >> 4) & 1; // half-wave
  const int l15  = lane & 15;
  const int row0 = blockIdx.x * MT;

  // ---- stage x tile: MT x D_MODEL fp32 -> bf16 in LDS (once per block) ----
  for (int idx = tid; idx < MT * (D_MODEL / 4); idx += 128) {
    const int r  = idx >> 9;        // D_MODEL/4 == 512
    const int c4 = idx & 511;
    const f32x4 f = *reinterpret_cast<const f32x4*>(
        x + (size_t)(row0 + r) * D_MODEL + (size_t)c4 * 4);
    unsigned short* p = ldsX + r * PITCH_X + c4 * 4;
    p[0] = f2bf(f.x); p[1] = f2bf(f.y); p[2] = f2bf(f.z); p[3] = f2bf(f.w);
  }

  float vpart[8];
#pragma unroll
  for (int r = 0; r < 8; ++r) vpart[r] = 0.f;

  __syncthreads();

  const int m_a = wave * 16 + l15;  // A-matrix row for this lane (local tile row)

  // outer loop over groups of NB N-tiles (64 w1 rows per group)
  for (int nto = 0; nto < D_MODEL / (16 * NB); ++nto) {
    v8f acc[NB];
#pragma unroll
    for (int t = 0; t < NB; ++t)
      acc[t] = (v8f){0.f, 0.f, 0.f, 0.f, 0.f, 0.f, 0.f, 0.f};

    // prefetch next group's w1 rows (gfx1250 global_prefetch_b8)
    if (nto + 1 < D_MODEL / (16 * NB) && tid < 16 * NB)
      __builtin_prefetch(w1 + (size_t)((nto + 1) * 16 * NB + tid) * D_MODEL, 0, 1);

    for (int kc = 0; kc < D_MODEL / KC; ++kc) {
      __syncthreads();
      // ---- stage w1 chunk: 16*NB rows x KC halves ----
      for (int idx = tid; idx < 16 * NB * (KC / 4); idx += 128) {
        const int r  = idx >> 6;    // KC/4 == 64
        const int c4 = idx & 63;
        const f32x4 f = *reinterpret_cast<const f32x4*>(
            w1 + (size_t)(nto * 16 * NB + r) * D_MODEL + (size_t)(kc * KC) +
            (size_t)c4 * 4);
        unsigned short* p = ldsW + r * PITCH_W + c4 * 4;
        p[0] = f2bf(f.x); p[1] = f2bf(f.y); p[2] = f2bf(f.z); p[3] = f2bf(f.w);
      }
      __syncthreads();

#pragma unroll
      for (int ktl = 0; ktl < KC / 32; ++ktl) {
        // ---- hoisted fragment loads: A + B[0..NB-1] in distinct registers
        V16BF A;
        {
          const unsigned short* ap =
              ldsX + m_a * PITCH_X + kc * KC + ktl * 32 + hi * 8;
          A.q[0] = *reinterpret_cast<const u32x4*>(ap);
          A.q[1] = *reinterpret_cast<const u32x4*>(ap + 16);
        }
        V16BF Bf[NB];
#pragma unroll
        for (int t = 0; t < NB; ++t) {
          // B 32x16: lanes 0-15 col=lane K=0..7,16..23; lanes 16-31 col K=8..15,24..31
          const unsigned short* bpp =
              ldsW + (t * 16 + l15) * PITCH_W + ktl * 32 + hi * 8;
          Bf[t].q[0] = *reinterpret_cast<const u32x4*>(bpp);
          Bf[t].q[1] = *reinterpret_cast<const u32x4*>(bpp + 16);
        }
        // ---- 4 independent WMMA accumulate chains sharing one A fragment
#pragma unroll
        for (int t = 0; t < NB; ++t)
          acc[t] = __builtin_amdgcn_wmma_f32_16x16x32_bf16(
              false, A.v, false, Bf[t].v, (short)0, acc[t], false, false);
      }
    }

    // ---- fused epilogue: bias + relu + dot(w2) into per-row partials ----
    // C layout: vgpr r -> (M = r + 8*hi, N = l15)
#pragma unroll
    for (int t = 0; t < NB; ++t) {
      const int ng  = (nto * NB + t) * 16 + l15;
      const float b1v = b1[ng];
      const float w2v = w2[ng];
#pragma unroll
      for (int r = 0; r < 8; ++r) {
        float h = acc[t][r] + b1v;
        h = h > 0.f ? h : 0.f;
        vpart[r] += h * w2v;
      }
    }
  }

  // reduce over the 16 lanes of each half-wave (N direction)
  const float b2v = b2[0];
#pragma unroll
  for (int r = 0; r < 8; ++r) {
    float s = vpart[r];
    s += __shfl_xor(s, 1, 32);
    s += __shfl_xor(s, 2, 32);
    s += __shfl_xor(s, 4, 32);
    s += __shfl_xor(s, 8, 32);
    if (l15 == 0) {
      const int row = row0 + wave * 16 + hi * 8 + r;
      v_out[row] = s + b2v;
    }
  }
}

// ---------------------------------------------------------------------------
// Kernel 2: continuant scan in complex128 via 2x2 matrix-product prefix scan.
// blockIdx.x==0 -> theta[0..n]; blockIdx.x==1 -> phi[0..n-1] (reversed scan)
// ---------------------------------------------------------------------------
struct cplx { double re, im; };
__device__ __forceinline__ cplx cmul(cplx a, cplx b) {
  return {a.re * b.re - a.im * b.im, a.re * b.im + a.im * b.re};
}
__device__ __forceinline__ cplx cadd(cplx a, cplx b) { return {a.re + b.re, a.im + b.im}; }
__device__ __forceinline__ cplx csub(cplx a, cplx b) { return {a.re - b.re, a.im - b.im}; }
struct mat2 { cplx a, b, c, d; };  // [[a,b],[c,d]]
__device__ __forceinline__ mat2 mmul(const mat2& X, const mat2& Y) {
  mat2 r;
  r.a = cadd(cmul(X.a, Y.a), cmul(X.b, Y.c));
  r.b = cadd(cmul(X.a, Y.b), cmul(X.b, Y.d));
  r.c = cadd(cmul(X.c, Y.a), cmul(X.d, Y.c));
  r.d = cadd(cmul(X.c, Y.b), cmul(X.d, Y.d));
  return r;
}

__global__ void __launch_bounds__(256)
scan_kernel(const float* __restrict__ v, cplx* __restrict__ theta,
            cplx* __restrict__ phi, int n) {
  __shared__ mat2 sm[256];
  const int dir = blockIdx.x;       // 0: forward theta, 1: reversed (phi)
  const int tid = threadIdx.x;
  const int CH  = n >> 8;           // elements per thread (8192/256 = 32)
  const int k0  = tid * CH;

  // local product of element matrices M(a)=[[a,-1],[1,0]], increasing index
  mat2 P = {{1, 0}, {0, 0}, {0, 0}, {1, 0}};
  for (int j = 0; j < CH; ++j) {
    const int k   = k0 + j;
    const int src = dir ? (n - 1 - k) : k;
    const cplx a  = {(double)v[src] - 2.0, -1.0};
    const cplx na = csub(cmul(a, P.a), P.c);
    const cplx nb = csub(cmul(a, P.b), P.d);
    P.c = P.a; P.d = P.b; P.a = na; P.b = nb;
  }
  sm[tid] = P;
  __syncthreads();
  if (tid == 0) {  // serial exclusive prefix over 256 block matrices
    mat2 R = {{1, 0}, {0, 0}, {0, 0}, {1, 0}};
    for (int t = 0; t < 256; ++t) {
      const mat2 tmp = sm[t];
      sm[t] = R;
      R = mmul(tmp, R);
    }
  }
  __syncthreads();

  const mat2 E = sm[tid];
  // s_k = (theta[k], theta[k-1]);  s_0 = (1,0)  ->  s_{k0} = (E.a, E.c)
  cplx t_prev = E.c, t_cur = E.a;
  if (dir == 0) {
    if (tid == 0) theta[0] = {1.0, 0.0};
    for (int j = 0; j < CH; ++j) {
      const int k  = k0 + j;
      const cplx a = {(double)v[k] - 2.0, -1.0};
      const cplx t_next = csub(cmul(a, t_cur), t_prev);
      theta[k + 1] = t_next;
      t_prev = t_cur; t_cur = t_next;
    }
  } else {
    // phi[k] = cont_rev[n-1-k];  cont_rev[m+1] written at phi[n-2-m]
    if (tid == 0) phi[n - 1] = {1.0, 0.0};
    for (int j = 0; j < CH; ++j) {
      const int m  = k0 + j;
      const cplx a = {(double)v[n - 1 - m] - 2.0, -1.0};
      const cplx t_next = csub(cmul(a, t_cur), t_prev);
      const int pidx = n - 2 - m;
      if (pidx >= 0) phi[pidx] = t_next;
      t_prev = t_cur; t_cur = t_next;
    }
  }
}

// ---------------------------------------------------------------------------
// Kernel 3: out[i,j] = Re(diag_i)*wp[j,0] + Im(diag_i)*wp[j,1] + bp[j]
// diag_i = theta[i]*phi[i]/(det+eps)   (complex128 -> complex64 -> f32 feats)
// ---------------------------------------------------------------------------
__global__ void __launch_bounds__(256)
proj_kernel(const cplx* __restrict__ theta, const cplx* __restrict__ phi,
            const float* __restrict__ wp, const float* __restrict__ bp,
            float* __restrict__ out, int n) {
  const int i = blockIdx.x;
  cplx det = theta[n];
  det.re += 1e-18;
  const cplx num = cmul(theta[i], phi[i]);
  const double d2 = det.re * det.re + det.im * det.im;
  const float fr = (float)((num.re * det.re + num.im * det.im) / d2);
  const float fi = (float)((num.im * det.re - num.re * det.im) / d2);
  float* orow = out + (size_t)i * D_MODEL;
  for (int j = threadIdx.x; j < D_MODEL; j += blockDim.x)
    orow[j] = fr * wp[2 * j] + fi * wp[2 * j + 1] + bp[j];
}

// ---------------------------------------------------------------------------
// Launch
// ---------------------------------------------------------------------------
extern "C" void kernel_launch(void* const* d_in, const int* in_sizes, int n_in,
                              void* d_out, int out_size, void* d_ws, size_t ws_size,
                              hipStream_t stream) {
  const float* x  = (const float*)d_in[0];
  const float* w1 = (const float*)d_in[1];
  const float* b1 = (const float*)d_in[2];
  const float* w2 = (const float*)d_in[3];
  const float* b2 = (const float*)d_in[4];
  const float* wp = (const float*)d_in[5];
  const float* bp = (const float*)d_in[6];

  const int n = in_sizes[0] / D_MODEL;  // 8192

  float* d_v     = (float*)d_ws;
  cplx*  d_theta = (cplx*)((char*)d_ws + 32768);
  cplx*  d_phi   = d_theta + (n + 1);

  gemm_v_kernel<<<n / MT, 128, GEMM_LDS_BYTES, stream>>>(x, w1, b1, w2, b2, d_v);
  scan_kernel<<<2, 256, 0, stream>>>(d_v, d_theta, d_phi, n);
  proj_kernel<<<n, 256, 0, stream>>>(d_theta, d_phi, wp, bp, (float*)d_out, n);
}
